// Model_55645596287354
// MI455X (gfx1250) — matrix operand order repaired and checked by reading, not compiled, not run on hardware
//
#include <hip/hip_runtime.h>

// Dims from the reference
#define VOCAB 50265
#define HD    512
#define ED    300
#define TT    64
#define BB    32
#define LENC  512

typedef __attribute__((ext_vector_type(16))) __bf16 v16bf;
typedef __attribute__((ext_vector_type(8)))  __bf16 v8bf;
typedef __attribute__((ext_vector_type(8)))  float  v8f;

// ---------------------------------------------------------------------------
// Init: h0 = init_states[0], ctx = 0
// ---------------------------------------------------------------------------
__global__ void init_state(const float* __restrict__ initS,
                           float* __restrict__ h0, float* __restrict__ ctx) {
  const int i = blockIdx.x * 256 + threadIdx.x;
  if (i < BB * HD) { h0[i] = initS[i]; ctx[i] = 0.f; }
}

// ---------------------------------------------------------------------------
// Embedding gather + x-part of W_red (parallel over all (t,b)):
// xred[t][b][e] = b_red[e] + emb[label[b][t]] @ W_red[0:E, e]
// ---------------------------------------------------------------------------
__global__ void embed_xred(const int* __restrict__ label, const float* __restrict__ emb,
                           const float* __restrict__ W_red, const float* __restrict__ b_red,
                           float* __restrict__ xred) {
  __shared__ float se[ED];
  const int tb = blockIdx.x;          // tb = t*BB + b
  const int t = tb >> 5, b = tb & 31;
  const int lbl = label[b * TT + t];
  for (int i = threadIdx.x; i < ED; i += 256) se[i] = emb[(size_t)lbl * ED + i];
  __syncthreads();
  for (int e = threadIdx.x; e < ED; e += 256) {
    float acc = b_red[e];
    for (int k = 0; k < ED; ++k) acc += se[k] * W_red[(size_t)k * ED + e];
    xred[(size_t)tb * ED + e] = acc;
  }
}

// ---------------------------------------------------------------------------
// W_log [H][V] f32  ->  Wt [V][H] bf16  (LDS tile transpose, one-time)
// ---------------------------------------------------------------------------
__global__ void wlog_transpose(const float* __restrict__ W, __bf16* __restrict__ Wt) {
  __shared__ float tile[64][65];
  const int v0 = blockIdx.x * 64, k0 = blockIdx.y * 64;
  const int tx = threadIdx.x & 63, ty = threadIdx.x >> 6;   // ty in 0..3
#pragma unroll
  for (int r = 0; r < 64; r += 4) {
    const int v = v0 + tx, k = k0 + ty + r;                 // k < 512 always
    tile[ty + r][tx] = (v < VOCAB) ? W[(size_t)k * VOCAB + v] : 0.f;
  }
  __syncthreads();
#pragma unroll
  for (int r = 0; r < 64; r += 4) {
    const int v = v0 + ty + r, k = k0 + tx;
    if (v < VOCAB) Wt[(size_t)v * HD + k] = (__bf16)tile[tx][ty + r];
  }
}

// ---------------------------------------------------------------------------
// One recurrence step. Block = batch row b (32 blocks x 512 threads).
// Computes: lstm_in -> GRU gates -> h_new -> attention A/B -> gate g ->
// context (next ctx) -> logit_in (stored bf16 for the WMMA GEMM).
// ---------------------------------------------------------------------------
__global__ void __launch_bounds__(HD)
step_kernel(const float* __restrict__ enc,   // [B,LENC,H]
            const float* __restrict__ dec,   // [B,T,H]
            const int*   __restrict__ emask, // [B,LENC]
            const int*   __restrict__ dmask, // [B,T]
            const float* __restrict__ W_red, // [E+H,E]
            const float* __restrict__ W_ih,  // [E,3H]
            const float* __restrict__ W_hh,  // [H,3H]
            const float* __restrict__ b_ih, const float* __restrict__ b_hh,
            const float* __restrict__ Wa, const float* __restrict__ ba,
            const float* __restrict__ Wb, const float* __restrict__ bb,
            const float* __restrict__ Wo, const float* __restrict__ bo,
            const float* __restrict__ W_cat, const float* __restrict__ b_cat,
            const float* __restrict__ xred,  // [T,B,E]
            const float* __restrict__ h_in,  // [B,H]
            float* __restrict__ h_out,       // [B,H]
            float* __restrict__ ctx,         // [B,H] (read early, overwritten late)
            __bf16* __restrict__ lin_bf,     // [T*B,H]
            int t) {
  __shared__ float s_ctx[HD], s_h[HD], s_q[HD], s_scores[LENC], s_red[HD];
  __shared__ float s_cA[HD], s_cB[HD], s_cx[HD], s_lin[ED + 4];
  const int b = blockIdx.x;
  const int tid = threadIdx.x;            // 512 threads

  s_ctx[tid] = ctx[b * HD + tid];
  s_h[tid]   = h_in[b * HD + tid];
  __syncthreads();

  // ---- phase 1: lstm_in[e] = xred + ctx @ W_red[E:,:] ----
  if (tid < ED) {
    float acc = xred[((size_t)t * BB + b) * ED + tid];
#pragma unroll 4
    for (int k = 0; k < HD; ++k) acc += s_ctx[k] * W_red[(size_t)(ED + k) * ED + tid];
    s_lin[tid] = acc;
  }
  __syncthreads();

  // ---- phase 2: GRU gates (torch order r,z,n), one thread per hidden j ----
  {
    const int j = tid;
    float gir = b_ih[j], giz = b_ih[HD + j], gin = b_ih[2 * HD + j];
    for (int e = 0; e < ED; ++e) {
      const float x = s_lin[e];
      const float* w = W_ih + (size_t)e * (3 * HD) + j;
      gir += x * w[0]; giz += x * w[HD]; gin += x * w[2 * HD];
    }
    float ghr = b_hh[j], ghz = b_hh[HD + j], ghn = b_hh[2 * HD + j];
#pragma unroll 4
    for (int k = 0; k < HD; ++k) {
      const float x = s_h[k];
      const float* w = W_hh + (size_t)k * (3 * HD) + j;
      ghr += x * w[0]; ghz += x * w[HD]; ghn += x * w[2 * HD];
    }
    const float r = 1.f / (1.f + __expf(-(gir + ghr)));
    const float z = 1.f / (1.f + __expf(-(giz + ghz)));
    const float n = tanhf(gin + r * ghn);
    const float hn = (1.f - z) * n + z * s_h[j];
    s_q[j] = hn;
    h_out[b * HD + j] = hn;
  }
  __syncthreads();

  // ---- phase 3: attention over encoder_outputs ----
  {
    const int l = tid;
    const float4* er = (const float4*)(enc + ((size_t)b * LENC + l) * HD);
    float acc = 0.f;
#pragma unroll 4
    for (int k = 0; k < HD / 4; ++k) {
      const float4 e4 = er[k];
      acc += s_q[4 * k] * e4.x + s_q[4 * k + 1] * e4.y +
             s_q[4 * k + 2] * e4.z + s_q[4 * k + 3] * e4.w;
    }
    if (emask[b * LENC + l] == 0) acc = -1e30f;
    s_scores[l] = acc;
  }
  __syncthreads();
  s_red[tid] = s_scores[tid];
  __syncthreads();
  for (int s = HD / 2; s > 0; s >>= 1) {
    if (tid < s) s_red[tid] = fmaxf(s_red[tid], s_red[tid + s]);
    __syncthreads();
  }
  const float mxA = s_red[0];
  __syncthreads();
  {
    const float p = __expf(s_scores[tid] - mxA);
    s_scores[tid] = p;
    s_red[tid] = p;
  }
  __syncthreads();
  for (int s = HD / 2; s > 0; s >>= 1) {
    if (tid < s) s_red[tid] += s_red[tid + s];
    __syncthreads();
  }
  const float invA = 1.f / s_red[0];
  __syncthreads();
  {
    const int h = tid;
    float acc = 0.f;
#pragma unroll 4
    for (int l = 0; l < LENC; ++l) acc += s_scores[l] * enc[((size_t)b * LENC + l) * HD + h];
    s_cA[h] = acc * invA;
  }
  __syncthreads();

  // ---- phase 4: attention over decoder_emb (L = 64) ----
  if (tid < TT) {
    const float4* dr = (const float4*)(dec + ((size_t)b * TT + tid) * HD);
    float acc = 0.f;
#pragma unroll 4
    for (int k = 0; k < HD / 4; ++k) {
      const float4 e4 = dr[k];
      acc += s_q[4 * k] * e4.x + s_q[4 * k + 1] * e4.y +
             s_q[4 * k + 2] * e4.z + s_q[4 * k + 3] * e4.w;
    }
    if (dmask[b * TT + tid] == 0) acc = -1e30f;
    s_scores[tid] = acc;
  }
  __syncthreads();
  if (tid == 0) {
    float mx = -3.4e38f;
    for (int l = 0; l < TT; ++l) mx = fmaxf(mx, s_scores[l]);
    float sum = 0.f;
    for (int l = 0; l < TT; ++l) { const float p = __expf(s_scores[l] - mx); s_scores[l] = p; sum += p; }
    s_red[0] = 1.f / sum;
  }
  __syncthreads();
  {
    const float invB = s_red[0];
    float acc = 0.f;
#pragma unroll 4
    for (int l = 0; l < TT; ++l) acc += s_scores[l] * dec[((size_t)b * TT + l) * HD + tid];
    s_cB[tid] = acc * invB;
  }
  __syncthreads();

  // ---- phase 5: mixing gate g, context = new ctx ----
  {
    const int j = tid;
    float gv = ba[j] + bb[j] + bo[j];
#pragma unroll 2
    for (int k = 0; k < HD; ++k)
      gv += s_cA[k] * Wa[(size_t)k * HD + j] + s_cB[k] * Wb[(size_t)k * HD + j] +
            s_q[k] * Wo[(size_t)k * HD + j];
    const float g = 1.f / (1.f + __expf(-gv));
    const float cx = g * s_cA[j] + (1.f - g) * s_cB[j];
    s_cx[j] = cx;
    ctx[b * HD + j] = cx;
  }
  __syncthreads();

  // ---- phase 6: logit_in = tanh([out, context] @ W_cat + b_cat), store bf16 ----
  {
    const int j = tid;
    float acc = b_cat[j];
#pragma unroll 4
    for (int k = 0; k < HD; ++k) acc += s_q[k] * W_cat[(size_t)k * HD + j];
#pragma unroll 4
    for (int k = 0; k < HD; ++k) acc += s_cx[k] * W_cat[(size_t)(HD + k) * HD + j];
    lin_bf[((size_t)t * BB + b) * HD + j] = (__bf16)tanhf(acc);
  }
}

// ---------------------------------------------------------------------------
// Batched logits GEMM: [M=T*B=2048] x [N=V] x [K=H=512], bf16 WMMA, f32 acc.
// Block = 8 waves = 4 m-groups x 2 n-groups; block tile = 128(M) x 128(N).
// Wave tile = 32(M) x 64(N): 2 M-tiles x 4 N-frags = 8 accumulators.
// Per k-step per wave: 4 A-loads + 8 B-loads (b128) feed 8 WMMAs.
// Only 16 M-groups sweep the 51 MB bf16 W_log image -> ~0.8 GB L2 traffic.
// A: logit_in [M][H] bf16 (row-major).  B: Wt [V][H] bf16 (pre-transposed).
// ---------------------------------------------------------------------------
__global__ void __launch_bounds__(256)
logits_wmma(const __bf16* __restrict__ A, const __bf16* __restrict__ Bt,
            const float* __restrict__ bias, float* __restrict__ out) {
  const int lane = threadIdx.x & 31;
  const int wave = threadIdx.x >> 5;                   // 0..7
  const int mTile0 = blockIdx.y * 8 + (wave & 3) * 2;  // first of 2 16-row tiles
  const int nBase  = blockIdx.x * 128 + (wave >> 2) * 64;
  const int half = lane >> 4;                          // 0/1
  const int mr = lane & 15;                            // row (A) / col (B,C)

  v8f acc[2][4] = {{v8f{}, v8f{}, v8f{}, v8f{}}, {v8f{}, v8f{}, v8f{}, v8f{}}};
  const __bf16* Arow0 = A + ((size_t)mTile0 * 16 + mr) * HD;
  const __bf16* Arow1 = Arow0 + (size_t)16 * HD;

  for (int kt = 0; kt < HD; kt += 32) {
    // A fragments: lane<16 -> K {kt..kt+7, kt+16..kt+23}; lane>=16 -> +8
    v16bf af0, af1;
    {
      const v8bf a00 = *(const v8bf*)(Arow0 + kt + half * 8);
      const v8bf a01 = *(const v8bf*)(Arow0 + kt + half * 8 + 16);
      const v8bf a10 = *(const v8bf*)(Arow1 + kt + half * 8);
      const v8bf a11 = *(const v8bf*)(Arow1 + kt + half * 8 + 16);
#pragma unroll
      for (int i = 0; i < 8; ++i) {
        af0[i] = a00[i]; af0[8 + i] = a01[i];
        af1[i] = a10[i]; af1[8 + i] = a11[i];
      }
    }
#pragma unroll
    for (int f = 0; f < 4; ++f) {
      int v = nBase + f * 16 + mr;
      if (v > VOCAB - 1) v = VOCAB - 1;                // clamp tail loads
      const __bf16* bp = Bt + (size_t)v * HD + kt + half * 8;
      const v8bf b0 = *(const v8bf*)(bp);
      const v8bf b1 = *(const v8bf*)(bp + 16);
      v16bf bf_;
#pragma unroll
      for (int i = 0; i < 8; ++i) { bf_[i] = b0[i]; bf_[8 + i] = b1[i]; }
      acc[0][f] = __builtin_amdgcn_wmma_f32_16x16x32_bf16(
          false, af0, false, bf_, (short)0, acc[0][f], false, false);
      acc[1][f] = __builtin_amdgcn_wmma_f32_16x16x32_bf16(
          false, af1, false, bf_, (short)0, acc[1][f], false, false);
    }
  }

  // C layout: element r -> M = tile*16 + half*8 + r ; N = lane&15
#pragma unroll
  for (int mi = 0; mi < 2; ++mi) {
#pragma unroll
    for (int f = 0; f < 4; ++f) {
      const int v = nBase + f * 16 + mr;
      if (v < VOCAB) {
        const float bv = bias[v];
#pragma unroll
        for (int r = 0; r < 8; ++r) {
          const int m = (mTile0 + mi) * 16 + half * 8 + r;  // m = t*B + b
          const int bi = m & 31, ti = m >> 5;
          out[((size_t)bi * TT + ti) * (size_t)VOCAB + v] = acc[mi][f][r] + bv;
        }
      }
    }
  }
}

// ---------------------------------------------------------------------------
extern "C" void kernel_launch(void* const* d_in, const int* in_sizes, int n_in,
                              void* d_out, int out_size, void* d_ws, size_t ws_size,
                              hipStream_t stream) {
  (void)in_sizes; (void)n_in; (void)out_size; (void)ws_size;
  const float* enc   = (const float*)d_in[0];
  const float* dec   = (const float*)d_in[1];
  const float* initS = (const float*)d_in[2];
  const int*   emask = (const int*)d_in[3];
  const int*   label = (const int*)d_in[4];
  const int*   dmask = (const int*)d_in[5];
  // d_in[6] = scheduled_probs (>=1 -> teacher forcing path, which is all the
  // reference implements)
  const float* emb   = (const float*)d_in[7];
  const float* W_red = (const float*)d_in[8];
  const float* b_red = (const float*)d_in[9];
  const float* W_ih  = (const float*)d_in[10];
  const float* W_hh  = (const float*)d_in[11];
  const float* b_ih  = (const float*)d_in[12];
  const float* b_hh  = (const float*)d_in[13];
  const float* Wa    = (const float*)d_in[14];
  const float* ba    = (const float*)d_in[15];
  const float* Wb    = (const float*)d_in[16];
  const float* bb    = (const float*)d_in[17];
  const float* Wo    = (const float*)d_in[18];
  const float* bo    = (const float*)d_in[19];
  const float* W_cat = (const float*)d_in[20];
  const float* b_cat = (const float*)d_in[21];
  const float* W_log = (const float*)d_in[22];
  const float* b_log = (const float*)d_in[23];
  float* out = (float*)d_out;

  // Workspace carve (~56 MB total)
  char* ws = (char*)d_ws;
  size_t off = 0;
  auto carve = [&](size_t bytes) -> void* {
    void* p = ws + off;
    off = (off + bytes + 255) & ~(size_t)255;
    return p;
  };
  __bf16* Wt    = (__bf16*)carve((size_t)VOCAB * HD * 2);  // 51.5 MB
  __bf16* linbf = (__bf16*)carve((size_t)TT * BB * HD * 2);
  float*  xred  = (float*)carve((size_t)TT * BB * ED * 4);
  float*  h0    = (float*)carve((size_t)BB * HD * 4);
  float*  h1    = (float*)carve((size_t)BB * HD * 4);
  float*  ctx   = (float*)carve((size_t)BB * HD * 4);

  // Independent parallel prep
  wlog_transpose<<<dim3((VOCAB + 63) / 64, HD / 64), 256, 0, stream>>>(W_log, Wt);
  init_state<<<(BB * HD + 255) / 256, 256, 0, stream>>>(initS, h0, ctx);
  embed_xred<<<TT * BB, 256, 0, stream>>>(label, emb, W_red, b_red, xred);

  // Sequential recurrence: one kernel per timestep (stream-ordered chain)
  for (int t = 0; t < TT; ++t) {
    float* hin  = (t & 1) ? h1 : h0;
    float* hout = (t & 1) ? h0 : h1;
    step_kernel<<<BB, HD, 0, stream>>>(enc, dec, emask, dmask, W_red, W_ih, W_hh,
                                       b_ih, b_hh, Wa, ba, Wb, bb, Wo, bo,
                                       W_cat, b_cat, xred, hin, hout, ctx, linbf, t);
  }

  // Batched vocabulary projection for all 2048 (t,b) rows with bf16 WMMA
  logits_wmma<<<dim3((VOCAB + 127) / 128, (TT * BB) / 128), 256, 0, stream>>>(
      linbf, Wt, b_log, out);
}
